// TransformerBlock_84464826843455
// MI455X (gfx1250) — matrix operand order repaired and checked by reading, not compiled, not run on hardware
//
#include <hip/hip_runtime.h>
#include <hip/hip_bf16.h>
#include <math.h>

typedef __attribute__((ext_vector_type(16))) __bf16          v16bf;
typedef __attribute__((ext_vector_type(8)))  float           v8f;
typedef unsigned short u16;
typedef __attribute__((ext_vector_type(8)))  unsigned short  u16x8; // 16 bytes

union FragB16 {
    v16bf v;
    u16x8 h[2];
};

__device__ __forceinline__ u16 f2bf(float f) {
    unsigned u = __float_as_uint(f);
    unsigned r = u + 0x7FFFu + ((u >> 16) & 1u); // round-to-nearest-even
    return (u16)(r >> 16);
}

__device__ __forceinline__ v8f zero8() {
    v8f v;
#pragma unroll
    for (int i = 0; i < 8; ++i) v[i] = 0.f;
    return v;
}

// ---------------- CDNA5 async global->LDS copy (ASYNCcnt-tracked DMA) -------
#define USE_ASYNC_LDS 1

__device__ __forceinline__ void async_copy16(void* lds_dst, const void* gsrc) {
#if USE_ASYNC_LDS
    unsigned lo = (unsigned)(uintptr_t)lds_dst; // LDS offset = low 32 bits
    unsigned long long ga = (unsigned long long)(uintptr_t)gsrc;
    asm volatile("global_load_async_to_lds_b128 %0, %1, off"
                 :: "v"(lo), "v"(ga) : "memory");
#else
    *(u16x8*)lds_dst = *(const u16x8*)gsrc;
#endif
}
__device__ __forceinline__ void async_wait() {
#if USE_ASYNC_LDS
    asm volatile("s_wait_asynccnt 0x0" ::: "memory");
#endif
}

// ------------- width-16 xor butterfly via v_permlane16_b32 (VALU, no LDS) ---
#define USE_PERMLANE 1

template <unsigned S0, unsigned S1>
__device__ __forceinline__ float permf(float v) {
#if USE_PERMLANE
    unsigned u = __float_as_uint(v);
    unsigned r = __builtin_amdgcn_permlane16(u, u, S0, S1, false, false);
    return __uint_as_float(r);
#else
    (void)S0; (void)S1;
    return v;
#endif
}
__device__ __forceinline__ float redmax16(float v) {
#if USE_PERMLANE
    v = fmaxf(v, permf<0x67452301u, 0xEFCDAB89u>(v)); // xor 1
    v = fmaxf(v, permf<0x54761032u, 0xDCFE98BAu>(v)); // xor 2
    v = fmaxf(v, permf<0x32107654u, 0xBA98FEDCu>(v)); // xor 4
    v = fmaxf(v, permf<0xFEDCBA98u, 0x76543210u>(v)); // xor 8
#else
    v = fmaxf(v, __shfl_xor(v, 1, 16));
    v = fmaxf(v, __shfl_xor(v, 2, 16));
    v = fmaxf(v, __shfl_xor(v, 4, 16));
    v = fmaxf(v, __shfl_xor(v, 8, 16));
#endif
    return v;
}
__device__ __forceinline__ float redsum16(float v) {
#if USE_PERMLANE
    v += permf<0x67452301u, 0xEFCDAB89u>(v);
    v += permf<0x54761032u, 0xDCFE98BAu>(v);
    v += permf<0x32107654u, 0xBA98FEDCu>(v);
    v += permf<0xFEDCBA98u, 0x76543210u>(v);
#else
    v += __shfl_xor(v, 1, 16);
    v += __shfl_xor(v, 2, 16);
    v += __shfl_xor(v, 4, 16);
    v += __shfl_xor(v, 8, 16);
#endif
    return v;
}

// ---------------------------------------------------------------- f32 -> bf16
__global__ void cvt_bf16_kernel(const float* __restrict__ src,
                                u16* __restrict__ dst, int n4) {
    int i = blockIdx.x * blockDim.x + threadIdx.x;
    if (i < n4) {
        float4 f = ((const float4*)src)[i];
        ushort4 o;
        o.x = f2bf(f.x); o.y = f2bf(f.y); o.z = f2bf(f.z); o.w = f2bf(f.w);
        ((ushort4*)dst)[i] = o;
    }
}

// -------------------------------- f32 (KxN) -> bf16 transposed (NxK), tiled
__global__ __launch_bounds__(256) void cvt_t_kernel(
    const float* __restrict__ src, u16* __restrict__ dst, int K, int N) {
    __shared__ u16 t[64][72]; // [n][k], padded
    const int k0 = blockIdx.y * 64, n0 = blockIdx.x * 64;
    const int tid = threadIdx.x;
    const int nn = (tid & 15) * 4, kr = tid >> 4;
#pragma unroll
    for (int p = 0; p < 4; ++p) {
        int k = kr + p * 16;
        float4 f = *(const float4*)(src + (size_t)(k0 + k) * N + n0 + nn);
        t[nn + 0][k] = f2bf(f.x);
        t[nn + 1][k] = f2bf(f.y);
        t[nn + 2][k] = f2bf(f.z);
        t[nn + 3][k] = f2bf(f.w);
    }
    __syncthreads();
    const int kk = (tid & 15) * 4, nr = tid >> 4;
#pragma unroll
    for (int p = 0; p < 4; ++p) {
        int n = nr + p * 16;
        ushort4 o;
        o.x = t[n][kk]; o.y = t[n][kk + 1]; o.z = t[n][kk + 2]; o.w = t[n][kk + 3];
        *(ushort4*)(dst + (size_t)(n0 + n) * K + k0 + kk) = o;
    }
}

// ------------------------------------------------------------ log(weights)
__global__ void logw_kernel(const float* __restrict__ w,
                            float* __restrict__ lw, int n) {
    int i = blockIdx.x * blockDim.x + threadIdx.x;
    if (i < n) lw[i] = logf(w[i] + 1e-20f);
}

// ---------------------------------------------------------------- WMMA GEMM
// C(MxN) = A(MxK,bf16) @ W^T(NxK,bf16, pre-transposed) + bias
// MODE 0: fp32 out   MODE 1: bf16 out   MODE 2: bf16 relu out
// MODE 3: bf16 out scattered as per-head V^T:  Vt[b][n][s] (n = h*64+d)
#define TK 32
#define LDS_S 40   // TK + 8 pad (u16 units)

template <int MODE>
__global__ __launch_bounds__(256) void gemm_bf16_kernel(
    const u16* __restrict__ A, const u16* __restrict__ Wt,
    const float* __restrict__ bias, int M, int N, int K,
    float* __restrict__ outF, u16* __restrict__ outB)
{
    __shared__ u16 As[2][128 * LDS_S];
    __shared__ u16 Bs[2][128 * LDS_S];

    const int m0 = blockIdx.y * 128, n0 = blockIdx.x * 128;
    const int tid = threadIdx.x, lane = tid & 31, wave = tid >> 5;
    const int wr = wave >> 2, wc = wave & 3;       // 2x4 wave grid -> 64x32 tiles
    const int lh = lane >> 4, ln = lane & 15;
    const int kloA = lh ? 8 : 0;                    // A-frag K interleave
    const int kloB = lh ? 8 : 0;                    // B-frag K interleave

    v8f acc[4][2];
#pragma unroll
    for (int i = 0; i < 4; ++i)
#pragma unroll
        for (int j = 0; j < 2; ++j) acc[i][j] = zero8();

    auto load_stage = [&](int ks, int buf) {
        const int k0 = ks * TK;
        // A tile: 128 rows x 32 k -> async DMA straight into LDS
#pragma unroll
        for (int i = 0; i < 2; ++i) {
            int c = tid + i * 256;
            int row = c >> 2, cc = (c & 3) * 8;
            async_copy16(&As[buf][row * LDS_S + cc],
                         A + (size_t)(m0 + row) * K + k0 + cc);
        }
        // B tile: 128 n-rows x 32 k from pre-transposed W^T -> async DMA
#pragma unroll
        for (int i = 0; i < 2; ++i) {
            int c = tid + i * 256;
            int row = c >> 2, cc = (c & 3) * 8;
            async_copy16(&Bs[buf][row * LDS_S + cc],
                         Wt + (size_t)(n0 + row) * K + k0 + cc);
        }
    };

    load_stage(0, 0);
    async_wait();
    __syncthreads();
    const int Ksteps = K / TK;
    for (int ks = 0; ks < Ksteps; ++ks) {
        const int buf = ks & 1;
        if (ks + 1 < Ksteps) load_stage(ks + 1, buf ^ 1);

        FragB16 bfr[2], afr[4];
#pragma unroll
        for (int nt = 0; nt < 2; ++nt) {
            int n = wc * 32 + nt * 16 + ln;
            bfr[nt].h[0] = *(const u16x8*)(&Bs[buf][n * LDS_S + kloB]);
            bfr[nt].h[1] = *(const u16x8*)(&Bs[buf][n * LDS_S + kloB + 16]);
        }
#pragma unroll
        for (int mt = 0; mt < 4; ++mt) {
            int row = wr * 64 + mt * 16 + ln;
            afr[mt].h[0] = *(const u16x8*)(&As[buf][row * LDS_S + kloA]);
            afr[mt].h[1] = *(const u16x8*)(&As[buf][row * LDS_S + kloA + 16]);
        }
#pragma unroll
        for (int mt = 0; mt < 4; ++mt)
#pragma unroll
            for (int nt = 0; nt < 2; ++nt)
                acc[mt][nt] = __builtin_amdgcn_wmma_f32_16x16x32_bf16(
                    false, afr[mt].v, false, bfr[nt].v, (short)0, acc[mt][nt],
                    false, false);
        async_wait();
        __syncthreads();
    }

#pragma unroll
    for (int mt = 0; mt < 4; ++mt)
#pragma unroll
        for (int nt = 0; nt < 2; ++nt)
#pragma unroll
            for (int r = 0; r < 8; ++r) {
                int m = m0 + wr * 64 + mt * 16 + r + (lh ? 8 : 0);
                int n = n0 + wc * 32 + nt * 16 + ln;
                float v = acc[mt][nt][r] + bias[n];
                if (MODE == 0) {
                    outF[(size_t)m * N + n] = v;
                } else if (MODE == 1) {
                    outB[(size_t)m * N + n] = f2bf(v);
                } else if (MODE == 2) {
                    outB[(size_t)m * N + n] = f2bf(v > 0.f ? v : 0.f);
                } else { // MODE 3: V^T per head
                    int b = m >> 11, s = m & 2047;
                    outB[(size_t)b * (1024 * 2048) + (size_t)n * 2048 + s] =
                        f2bf(v);
                }
            }
}

// ---------------------------------------------------------------- attention
// Flash-style. One wave = 16 query rows, full d_k=64 accumulator.
// 64-key K/V tiles staged into LDS once per workgroup (8 waves share),
// double-buffered via async DMA.
#define KS 72   // K/V tile LDS stride (64 + 8 pad), 144B rows (16B aligned)

__global__ __launch_bounds__(256) void attn_kernel(
    const u16* __restrict__ Qb, const u16* __restrict__ Kb,
    const u16* __restrict__ Vt, const float* __restrict__ lwbuf,
    u16* __restrict__ outB)
{
    __shared__ u16 Kst[2][64 * KS];   // [key][d]
    __shared__ u16 Vst[2][64 * KS];   // [d][key]
    __shared__ u16 Pl[8][16 * 64];    // per-wave P tile (16 q x 64 keys)

    const int tid = threadIdx.x, lane = tid & 31, wave = tid >> 5;
    const int lh = lane >> 4, ln = lane & 15;
    const int kloA = lh ? 8 : 0, kloB = lh ? 8 : 0;
    const int bh = blockIdx.y, b = bh >> 4, h = bh & 15;
    const int q0 = blockIdx.x * 128 + wave * 16;

    auto stage = [&](int kb, int buf) {
        // K tile: 64 keys x 64 d (512 x 16B chunks, 2 per thread)
        // V tile: 64 d x 64 keys (512 x 16B chunks, 2 per thread)
#pragma unroll
        for (int i = 0; i < 2; ++i) {
            int c = tid + i * 256;
            int row = c >> 3, cc = (c & 7) * 8;
            async_copy16(&Kst[buf][row * KS + cc],
                         Kb + (size_t)(b * 2048 + kb + row) * 1024 + h * 64 + cc);
            async_copy16(&Vst[buf][row * KS + cc],
                         Vt + (size_t)(b * 1024 + h * 64 + row) * 2048 + kb + cc);
        }
    };

    // Q fragments (two K=32 slabs of d_k=64)
    FragB16 qf[2];
    {
        const u16* qp = Qb + (size_t)(b * 2048 + q0 + ln) * 1024 + h * 64;
#pragma unroll
        for (int kk = 0; kk < 2; ++kk) {
            qf[kk].h[0] = *(const u16x8*)(qp + kk * 32 + kloA);
            qf[kk].h[1] = *(const u16x8*)(qp + kk * 32 + kloA + 16);
        }
    }

    v8f acc[4];
#pragma unroll
    for (int i = 0; i < 4; ++i) acc[i] = zero8();
    float mrow[8], lrow[8];
#pragma unroll
    for (int r = 0; r < 8; ++r) { mrow[r] = -1e30f; lrow[r] = 0.f; }

    const float scale = 0.125f; // 1/sqrt(64)

    stage(0, 0);
    async_wait();
    __syncthreads();

    for (int kb = 0; kb < 2048; kb += 64) {
        const int buf = (kb >> 6) & 1;
        if (kb + 64 < 2048) stage(kb + 64, buf ^ 1);

        v8f sc[4];
        float lw[4];
#pragma unroll
        for (int nt = 0; nt < 4; ++nt) {
            sc[nt] = zero8();
            const u16* kp = &Kst[buf][(nt * 16 + ln) * KS];
#pragma unroll
            for (int kk = 0; kk < 2; ++kk) {
                FragB16 kf;
                kf.h[0] = *(const u16x8*)(kp + kk * 32 + kloB);
                kf.h[1] = *(const u16x8*)(kp + kk * 32 + kloB + 16);
                sc[nt] = __builtin_amdgcn_wmma_f32_16x16x32_bf16(
                    false, qf[kk].v, false, kf.v, (short)0, sc[nt], false, false);
            }
            lw[nt] = lwbuf[b * 2048 + kb + nt * 16 + ln];
        }
#pragma unroll
        for (int nt = 0; nt < 4; ++nt)
#pragma unroll
            for (int r = 0; r < 8; ++r) sc[nt][r] = sc[nt][r] * scale + lw[nt];

        // online softmax (rows in vgpr index; cols across 16-lane halves)
#pragma unroll
        for (int r = 0; r < 8; ++r) {
            float v = fmaxf(fmaxf(sc[0][r], sc[1][r]), fmaxf(sc[2][r], sc[3][r]));
            v = redmax16(v);
            float mn = fmaxf(mrow[r], v);
            float corr = __expf(mrow[r] - mn);
            mrow[r] = mn;
            float ps = 0.f;
#pragma unroll
            for (int nt = 0; nt < 4; ++nt) {
                float p = __expf(sc[nt][r] - mn);
                sc[nt][r] = p;
                ps += p;
            }
            ps = redsum16(ps);
            lrow[r] = lrow[r] * corr + ps;
#pragma unroll
            for (int dt = 0; dt < 4; ++dt) acc[dt][r] *= corr;
        }

        // C-layout -> A-fragment layout via per-wave LDS round trip
#pragma unroll
        for (int nt = 0; nt < 4; ++nt)
#pragma unroll
            for (int r = 0; r < 8; ++r)
                Pl[wave][(r + lh * 8) * 64 + nt * 16 + ln] = f2bf(sc[nt][r]);
        FragB16 pf[2];
#pragma unroll
        for (int kk = 0; kk < 2; ++kk) {
            pf[kk].h[0] = *(const u16x8*)(&Pl[wave][ln * 64 + kk * 32 + kloA]);
            pf[kk].h[1] = *(const u16x8*)(&Pl[wave][ln * 64 + kk * 32 + kloA + 16]);
        }

#pragma unroll
        for (int dt = 0; dt < 4; ++dt) {
            const u16* vp = &Vst[buf][(dt * 16 + ln) * KS];
#pragma unroll
            for (int kk = 0; kk < 2; ++kk) {
                FragB16 vf;
                vf.h[0] = *(const u16x8*)(vp + kk * 32 + kloB);
                vf.h[1] = *(const u16x8*)(vp + kk * 32 + kloB + 16);
                acc[dt] = __builtin_amdgcn_wmma_f32_16x16x32_bf16(
                    false, pf[kk].v, false, vf.v, (short)0, acc[dt], false, false);
            }
        }
        async_wait();
        __syncthreads();
    }

#pragma unroll
    for (int dt = 0; dt < 4; ++dt)
#pragma unroll
        for (int r = 0; r < 8; ++r) {
            int q = q0 + r + lh * 8;
            int d = dt * 16 + ln;
            float v = acc[dt][r] / lrow[r];
            outB[(size_t)(b * 2048 + q) * 1024 + h * 64 + d] = f2bf(v);
        }
}

// ---------------------------------------------------------------- residual+LN
__device__ __forceinline__ float block_sum256(float s, volatile float* red) {
    int tid = threadIdx.x, lane = tid & 31, wave = tid >> 5;
#pragma unroll
    for (int m = 1; m < 32; m <<= 1) s += __shfl_xor(s, m, 32);
    __syncthreads();
    if (lane == 0) red[wave] = s;
    __syncthreads();
    float t = 0.f;
#pragma unroll
    for (int i = 0; i < 8; ++i) t += red[i];
    return t;
}

__global__ __launch_bounds__(256) void add_ln_kernel(
    const float* __restrict__ A, const float* __restrict__ Bv,
    const float* __restrict__ gamma, const float* __restrict__ beta,
    float* __restrict__ outF, u16* __restrict__ outB)
{
    __shared__ float red[8];
    const int row = blockIdx.x, tid = threadIdx.x;
    const float* a = A + (size_t)row * 1024;
    const float* bb = Bv + (size_t)row * 1024;
    float v[4];
    float s = 0.f;
#pragma unroll
    for (int i = 0; i < 4; ++i) {
        v[i] = a[tid + i * 256] + bb[tid + i * 256];
        s += v[i];
    }
    float mean = block_sum256(s, red) * (1.f / 1024.f);
    float d2 = 0.f;
#pragma unroll
    for (int i = 0; i < 4; ++i) {
        float d = v[i] - mean;
        d2 += d * d;
    }
    float var = block_sum256(d2, red) * (1.f / 1024.f);
    float rstd = rsqrtf(var + 1e-3f);
#pragma unroll
    for (int i = 0; i < 4; ++i) {
        int col = tid + i * 256;
        float o = (v[i] - mean) * rstd * gamma[col] + beta[col];
        if (outF) outF[(size_t)row * 1024 + col] = o;
        if (outB) outB[(size_t)row * 1024 + col] = f2bf(o);
    }
}

// ---------------------------------------------------------------- launcher
extern "C" void kernel_launch(void* const* d_in, const int* in_sizes, int n_in,
                              void* d_out, int out_size, void* d_ws,
                              size_t ws_size, hipStream_t stream)
{
    const float* x   = (const float*)d_in[0];
    const float* wts = (const float*)d_in[1];
    const float* Wq  = (const float*)d_in[2];
    const float* bq  = (const float*)d_in[3];
    const float* Wk  = (const float*)d_in[4];
    const float* bk  = (const float*)d_in[5];
    const float* Wv  = (const float*)d_in[6];
    const float* bv  = (const float*)d_in[7];
    const float* Wo  = (const float*)d_in[8];
    const float* bo  = (const float*)d_in[9];
    const float* W1  = (const float*)d_in[10];
    const float* b1  = (const float*)d_in[11];
    const float* W2  = (const float*)d_in[12];
    const float* b2  = (const float*)d_in[13];
    const float* g1  = (const float*)d_in[14];
    const float* be1 = (const float*)d_in[15];
    const float* g2  = (const float*)d_in[16];
    const float* be2 = (const float*)d_in[17];

    char* ws = (char*)d_ws;
    size_t off = 0;
    auto alloc = [&](size_t bytes) -> void* {
        void* p = (void*)(ws + off);
        off += (bytes + 255) & ~(size_t)255;
        return p;
    };
    const size_t M = 4096; // B*S
    u16* xb   = (u16*)alloc(M * 1024 * 2);
    u16* Wqt  = (u16*)alloc(1024ull * 1024 * 2);  // all weights stored N x K
    u16* Wkt  = (u16*)alloc(1024ull * 1024 * 2);
    u16* Wvt  = (u16*)alloc(1024ull * 1024 * 2);
    u16* Wot  = (u16*)alloc(1024ull * 1024 * 2);
    u16* W1t  = (u16*)alloc(1024ull * 4096 * 2);
    u16* W2t  = (u16*)alloc(4096ull * 1024 * 2);
    u16* Qb   = (u16*)alloc(M * 1024 * 2);
    u16* Kb2  = (u16*)alloc(M * 1024 * 2);
    u16* Vtb  = (u16*)alloc(M * 1024 * 2);
    u16* attb = (u16*)alloc(M * 1024 * 2);
    float* attO = (float*)alloc(M * 1024 * 4);
    float* hF   = (float*)alloc(M * 1024 * 4);
    u16* hB     = (u16*)alloc(M * 1024 * 2);
    u16* f1b    = (u16*)alloc(M * 4096 * 2);
    float* lwb  = (float*)alloc(M * 4);  // log(weights + 1e-20), B*S
    float* f2F  = attO; // reuse: attO consumed by LN1 before FFN2 writes

    {
        int n4 = (int)(M * 1024 / 4);
        cvt_bf16_kernel<<<(n4 + 255) / 256, 256, 0, stream>>>(x, xb, n4);
    }
    dim3 blk(256);
    // transpose-convert weights once: (KxN f32) -> (NxK bf16)
    cvt_t_kernel<<<dim3(16, 16), blk, 0, stream>>>(Wq, Wqt, 1024, 1024);
    cvt_t_kernel<<<dim3(16, 16), blk, 0, stream>>>(Wk, Wkt, 1024, 1024);
    cvt_t_kernel<<<dim3(16, 16), blk, 0, stream>>>(Wv, Wvt, 1024, 1024);
    cvt_t_kernel<<<dim3(16, 16), blk, 0, stream>>>(Wo, Wot, 1024, 1024);
    cvt_t_kernel<<<dim3(64, 16), blk, 0, stream>>>(W1, W1t, 1024, 4096);
    cvt_t_kernel<<<dim3(16, 64), blk, 0, stream>>>(W2, W2t, 4096, 1024);
    logw_kernel<<<(int)(M + 255) / 256, 256, 0, stream>>>(wts, lwb, (int)M);

    // Q, K projections -> bf16; V projection -> per-head transposed bf16
    gemm_bf16_kernel<1><<<dim3(8, 32), blk, 0, stream>>>(xb, Wqt, bq, 4096, 1024, 1024, nullptr, Qb);
    gemm_bf16_kernel<1><<<dim3(8, 32), blk, 0, stream>>>(xb, Wkt, bk, 4096, 1024, 1024, nullptr, Kb2);
    gemm_bf16_kernel<3><<<dim3(8, 32), blk, 0, stream>>>(xb, Wvt, bv, 4096, 1024, 1024, nullptr, Vtb);

    attn_kernel<<<dim3(16, 32), blk, 0, stream>>>(Qb, Kb2, Vtb, lwb, attb);

    // O projection -> fp32
    gemm_bf16_kernel<0><<<dim3(8, 32), blk, 0, stream>>>(attb, Wot, bo, 4096, 1024, 1024, attO, nullptr);
    // h = LN(x + attn_out)
    add_ln_kernel<<<4096, blk, 0, stream>>>(x, attO, g1, be1, hF, hB);
    // FFN1 (relu) -> bf16
    gemm_bf16_kernel<2><<<dim3(32, 32), blk, 0, stream>>>(hB, W1t, b1, 4096, 4096, 1024, nullptr, f1b);
    // FFN2 -> fp32
    gemm_bf16_kernel<0><<<dim3(8, 32), blk, 0, stream>>>(f1b, W2t, b2, 4096, 1024, 4096, f2F, nullptr);
    // out = LN(h + ffn)
    add_ln_kernel<<<4096, blk, 0, stream>>>(hF, f2F, g2, be2, (float*)d_out, nullptr);
}
